// DecoderRNN_57913339020019
// MI455X (gfx1250) — compile-verified, matrix operand order checked by reading, not run on hardware
//
#include <hip/hip_runtime.h>
#include <hip/hip_bf16.h>
#include <math.h>

#define B_ 64
#define S_ 512
#define H_ 1024
#define V_ 50257

typedef __attribute__((ext_vector_type(2)))  float  v2f;
typedef __attribute__((ext_vector_type(8)))  float  v8f;
typedef __attribute__((ext_vector_type(16))) __bf16 v16bf;

union Frag16 { uint4 q[2]; v16bf v; };

static __device__ __forceinline__ unsigned short f2bf_raw(float f) {
  unsigned u = __builtin_bit_cast(unsigned, f);
  unsigned r = u + 0x7fffu + ((u >> 16) & 1u);   // round-to-nearest-even
  return (unsigned short)(r >> 16);
}

// ---------------------------------------------------------------------------
// Convert enc-half of attn_w ([H, 2H] fp32, columns H..2H-1) to bf16 [H][H]
// ---------------------------------------------------------------------------
__global__ __launch_bounds__(256) void prep_wbf_kernel(
    const float* __restrict__ attn_w, unsigned short* __restrict__ wBf) {
  int n = blockIdx.x;                       // output row (H of them)
  for (int k = threadIdx.x; k < H_; k += 256)
    wBf[(size_t)n * H_ + k] = f2bf_raw(attn_w[(size_t)n * 2 * H_ + H_ + k]);
}

// ---------------------------------------------------------------------------
// Embedding gather -> xbuf[:, 0:H]
// ---------------------------------------------------------------------------
__global__ __launch_bounds__(256) void embed_kernel(
    const int* __restrict__ ids, const float* __restrict__ emb,
    float* __restrict__ xbuf) {
  int b = blockIdx.x;
  const float* src = emb + (size_t)ids[b] * H_;
  float* dst = xbuf + (size_t)b * 2 * H_;
  for (int j = threadIdx.x; j < H_; j += 256) dst[j] = src[j];
}

// ---------------------------------------------------------------------------
// Generic fp32 WMMA GEMM: C[M,N] = A[M,K] * B[N,K]^T + bias[N]
// One wave per 16x32 output tile (two 16x16 WMMA tiles, two accumulator
// chains). K-loop of v_wmma_f32_16x16x4_f32. OOB columns: load addresses are
// clamped (garbage only pollutes unstored accumulator columns), stores are
// guarded. M must be a multiple of 16 (M=64 everywhere here).
// ---------------------------------------------------------------------------
__global__ __launch_bounds__(32) void gemm_f32_wmma_kernel(
    const float* __restrict__ A, int lda,
    const float* __restrict__ Bm, int ldb,
    const float* __restrict__ bias,
    float* __restrict__ C, int ldc,
    int N, int K) {
  const int lane = threadIdx.x;
  const int half = lane >> 4;         // 0: K pair {0,1} / rows M..M+7 ; 1: {2,3} / M+8..
  const int l16  = lane & 15;
  const int m0 = blockIdx.y * 16;
  const int n0 = blockIdx.x * 32;
  const int nA = n0 + l16;
  const int nB = n0 + 16 + l16;
  const bool okA = (nA < N), okB = (nB < N);
  const int nlA = okA ? nA : (N - 1);       // clamped, always in-bounds
  const int nlB = okB ? nB : (N - 1);

  const float* __restrict__ Arow = A  + (size_t)(m0 + l16) * lda + half * 2;
  const float* __restrict__ Br0  = Bm + (size_t)nlA * ldb + half * 2;
  const float* __restrict__ Br1  = Bm + (size_t)nlB * ldb + half * 2;

  v8f acc0 = {0.f, 0.f, 0.f, 0.f, 0.f, 0.f, 0.f, 0.f};
  v8f acc1 = {0.f, 0.f, 0.f, 0.f, 0.f, 0.f, 0.f, 0.f};
  for (int k0 = 0; k0 < K; k0 += 4) {
    const v2f a  = *(const v2f*)(Arow + k0);
    const v2f b0 = *(const v2f*)(Br0 + k0);
    const v2f b1 = *(const v2f*)(Br1 + k0);
    acc0 = __builtin_amdgcn_wmma_f32_16x16x4_f32(
        false, a, false, b0, (short)0, acc0, false, false);
    acc1 = __builtin_amdgcn_wmma_f32_16x16x4_f32(
        false, a, false, b1, (short)0, acc1, false, false);
  }
  const int mbase = m0 + half * 8;
  if (okA) {
    const float bv = bias ? bias[nA] : 0.f;
#pragma unroll
    for (int r = 0; r < 8; ++r)
      C[(size_t)(mbase + r) * ldc + nA] = acc0[r] + bv;
  }
  if (okB) {
    const float bv = bias ? bias[nB] : 0.f;
#pragma unroll
    for (int r = 0; r < 8; ++r)
      C[(size_t)(mbase + r) * ldc + nB] = acc1[r] + bv;
  }
}

// ---------------------------------------------------------------------------
// Fused attention scores: scores[b,s] = sum_h v_w[h]*tanh(enc@We^T + hW)[b,s,h]
// Block = 16 rows (one b, 16 consecutive s), 4 waves. enc tile converted to
// bf16 in LDS once; per wave: 16 n-tiles x 32 K-steps of wmma_f32_16x16x32_bf16.
// ---------------------------------------------------------------------------
__global__ __launch_bounds__(128) void attn_scores_kernel(
    const float* __restrict__ enc,            // [B,S,H]
    const unsigned short* __restrict__ wBf,   // [H][H] bf16
    const float* __restrict__ hW,             // [B,H]  (h@Wh^T + attn_b)
    const float* __restrict__ vw,             // [H]
    float* __restrict__ scores) {             // [B,S]
  __shared__ __align__(16) unsigned short At[16][H_];  // 32 KB bf16 enc tile
  __shared__ float partial[4][16];

  const int blk = blockIdx.x;          // B*S/16 blocks
  const int b   = blk >> 5;            // S/16 == 32 tiles per b
  const int s0  = (blk & 31) * 16;
  const int tid = threadIdx.x;

  const float* __restrict__ encBase = enc + ((size_t)b * S_ + s0) * H_;
  for (int i = tid; i < 16 * H_; i += 128)
    At[i >> 10][i & (H_ - 1)] = f2bf_raw(encBase[i]);
  __syncthreads();

  const int wave = tid >> 5, lane = tid & 31;
  const int half = lane >> 4, l16 = lane & 15;

  float part[8];
#pragma unroll
  for (int r = 0; r < 8; ++r) part[r] = 0.f;

  const unsigned short* __restrict__ Ap = &At[l16][0];

  for (int nt = wave; nt < H_ / 16; nt += 4) {
    const int n = nt * 16 + l16;
    const unsigned short* __restrict__ Bp = wBf + (size_t)n * H_;
    v8f acc = {0.f, 0.f, 0.f, 0.f, 0.f, 0.f, 0.f, 0.f};
    for (int k0 = 0; k0 < H_; k0 += 32) {
      const int koff = k0 + half * 8;
      Frag16 fa, fb;
      fa.q[0] = *(const uint4*)(Ap + koff);
      fa.q[1] = *(const uint4*)(Ap + koff + 16);
      fb.q[0] = *(const uint4*)(Bp + koff);
      fb.q[1] = *(const uint4*)(Bp + koff + 16);
      acc = __builtin_amdgcn_wmma_f32_16x16x32_bf16(
          false, fa.v, false, fb.v, (short)0, acc, false, false);
    }
    const float hwv = hW[(size_t)b * H_ + n];
    const float vv  = vw[n];
#pragma unroll
    for (int r = 0; r < 8; ++r) part[r] += vv * tanhf(acc[r] + hwv);
  }

  // reduce over the 16 lanes of each half (the N dimension)
#pragma unroll
  for (int r = 0; r < 8; ++r) {
    float v = part[r];
    v += __shfl_xor(v, 1, 32);
    v += __shfl_xor(v, 2, 32);
    v += __shfl_xor(v, 4, 32);
    v += __shfl_xor(v, 8, 32);
    part[r] = v;
  }
  if (l16 == 0) {
#pragma unroll
    for (int r = 0; r < 8; ++r) partial[wave][half * 8 + r] = part[r];
  }
  __syncthreads();
  if (tid < 16)
    scores[(size_t)b * S_ + s0 + tid] =
        partial[0][tid] + partial[1][tid] + partial[2][tid] + partial[3][tid];
}

// ---------------------------------------------------------------------------
// Softmax over S per batch row
// ---------------------------------------------------------------------------
__global__ __launch_bounds__(256) void softmax_kernel(
    const float* __restrict__ scores, float* __restrict__ attn) {
  __shared__ float red[256];
  const int b = blockIdx.x, tid = threadIdx.x;
  const float* sr = scores + (size_t)b * S_;
  float x0 = sr[tid], x1 = sr[tid + 256];
  red[tid] = fmaxf(x0, x1);
  __syncthreads();
  for (int s = 128; s > 0; s >>= 1) {
    if (tid < s) red[tid] = fmaxf(red[tid], red[tid + s]);
    __syncthreads();
  }
  const float mx = red[0];
  __syncthreads();
  const float e0 = expf(x0 - mx), e1 = expf(x1 - mx);
  red[tid] = e0 + e1;
  __syncthreads();
  for (int s = 128; s > 0; s >>= 1) {
    if (tid < s) red[tid] += red[tid + s];
    __syncthreads();
  }
  const float inv = 1.f / red[0];
  attn[(size_t)b * S_ + tid]       = e0 * inv;
  attn[(size_t)b * S_ + tid + 256] = e1 * inv;
}

// ---------------------------------------------------------------------------
// context[b,h] = sum_s attn[b,s]*enc[b,s,h]  -> xbuf[:, H:2H]
// ---------------------------------------------------------------------------
__global__ __launch_bounds__(256) void context_kernel(
    const float* __restrict__ attn, const float* __restrict__ enc,
    float* __restrict__ xbuf) {
  __shared__ float aw[S_];
  const int b = blockIdx.x;
  const int h = blockIdx.y * 256 + threadIdx.x;
  for (int i = threadIdx.x; i < S_; i += 256) aw[i] = attn[(size_t)b * S_ + i];
  __syncthreads();
  const float* __restrict__ e = enc + (size_t)b * S_ * H_ + h;
  float acc = 0.f;
  for (int s = 0; s < S_; ++s) acc += aw[s] * e[(size_t)s * H_];
  xbuf[(size_t)b * 2 * H_ + H_ + h] = acc;
}

// ---------------------------------------------------------------------------
// GRU gate combine (PyTorch r,z,n order)
// ---------------------------------------------------------------------------
__global__ __launch_bounds__(256) void gru_gate_kernel(
    const float* __restrict__ gi, const float* __restrict__ gh,
    const float* __restrict__ h, float* __restrict__ h_new) {
  const int b = blockIdx.x;
  const int j = blockIdx.y * 256 + threadIdx.x;
  const size_t o = (size_t)b * 3 * H_;
  const float ir = gi[o + j],           hr = gh[o + j];
  const float iz = gi[o + H_ + j],      hz = gh[o + H_ + j];
  const float in = gi[o + 2 * H_ + j],  hn = gh[o + 2 * H_ + j];
  const float r = 1.f / (1.f + expf(-(ir + hr)));
  const float z = 1.f / (1.f + expf(-(iz + hz)));
  const float n = tanhf(in + r * hn);
  h_new[(size_t)b * H_ + j] = (1.f - z) * n + z * h[(size_t)b * H_ + j];
}

// ---------------------------------------------------------------------------
// Launch
// ---------------------------------------------------------------------------
extern "C" void kernel_launch(void* const* d_in, const int* in_sizes, int n_in,
                              void* d_out, int out_size, void* d_ws, size_t ws_size,
                              hipStream_t stream) {
  const int*   ids    = (const int*)  d_in[0];
  const float* hidden = (const float*)d_in[1];   // [1,B,H] == [B,H]
  const float* enc    = (const float*)d_in[2];   // [B,S,H]
  const float* emb    = (const float*)d_in[3];   // [V,H]
  const float* attn_w = (const float*)d_in[4];   // [H,2H]
  const float* attn_b = (const float*)d_in[5];   // [H]
  const float* v_w    = (const float*)d_in[6];   // [H]
  const float* w_ih   = (const float*)d_in[7];   // [3H,2H]
  const float* w_hh   = (const float*)d_in[8];   // [3H,H]
  const float* b_ih   = (const float*)d_in[9];   // [3H]
  const float* b_hh   = (const float*)d_in[10];  // [3H]
  const float* out_w  = (const float*)d_in[11];  // [V,H]
  const float* out_b  = (const float*)d_in[12];  // [V]

  // d_out layout: pred [B,V] | h_new [B,H] | attn [B,S]
  float* pred     = (float*)d_out;
  float* hnew_out = pred + (size_t)B_ * V_;
  float* attn_out = hnew_out + (size_t)B_ * H_;

  // workspace layout
  char* ws = (char*)d_ws;
  unsigned short* wBf = (unsigned short*)ws;              ws += (size_t)H_ * H_ * sizeof(unsigned short);
  float* scores = (float*)ws;                             ws += (size_t)B_ * S_ * sizeof(float);
  float* hW     = (float*)ws;                             ws += (size_t)B_ * H_ * sizeof(float);
  float* xbuf   = (float*)ws;                             ws += (size_t)B_ * 2 * H_ * sizeof(float);
  float* gi     = (float*)ws;                             ws += (size_t)B_ * 3 * H_ * sizeof(float);
  float* gh     = (float*)ws;                             ws += (size_t)B_ * 3 * H_ * sizeof(float);

  // 1) bf16 weights for enc-half of attn_w
  prep_wbf_kernel<<<H_, 256, 0, stream>>>(attn_w, wBf);
  // 2) embedding gather into xbuf[:, :H]
  embed_kernel<<<B_, 256, 0, stream>>>(ids, emb, xbuf);
  // 3) hW = h @ Wh^T + attn_b (f32 WMMA)
  gemm_f32_wmma_kernel<<<dim3(H_ / 32, B_ / 16), 32, 0, stream>>>(
      hidden, H_, attn_w, 2 * H_, attn_b, hW, H_, H_, H_);
  // 4) fused scores (bf16 WMMA + tanh + v-reduction)
  attn_scores_kernel<<<B_ * S_ / 16, 128, 0, stream>>>(enc, wBf, hW, v_w, scores);
  // 5) softmax -> attn output
  softmax_kernel<<<B_, 256, 0, stream>>>(scores, attn_out);
  // 6) context -> xbuf[:, H:2H]
  context_kernel<<<dim3(B_, H_ / 256), 256, 0, stream>>>(attn_out, enc, xbuf);
  // 7) GRU input/hidden GEMMs (f32 WMMA)
  gemm_f32_wmma_kernel<<<dim3(3 * H_ / 32, B_ / 16), 32, 0, stream>>>(
      xbuf, 2 * H_, w_ih, 2 * H_, b_ih, gi, 3 * H_, 3 * H_, 2 * H_);
  gemm_f32_wmma_kernel<<<dim3(3 * H_ / 32, B_ / 16), 32, 0, stream>>>(
      hidden, H_, w_hh, H_, b_hh, gh, 3 * H_, 3 * H_, H_);
  // 8) gates -> h_new output
  gru_gate_kernel<<<dim3(B_, H_ / 256), 256, 0, stream>>>(gi, gh, hidden, hnew_out);
  // 9) prediction = h_new @ out_w^T + out_b (f32 WMMA, N guarded)
  gemm_f32_wmma_kernel<<<dim3((V_ + 31) / 32, B_ / 16), 32, 0, stream>>>(
      hnew_out, H_, out_w, H_, out_b, pred, V_, V_, H_);
}
